// NASActor_84954453114978
// MI455X (gfx1250) — matrix operand order repaired and checked by reading, not compiled, not run on hardware
//
#include <hip/hip_runtime.h>

// ---------------------------------------------------------------------------
// NAS actor forward on gfx1250 (MI455X): bf16 WMMA GEMMs with f32 accumulate.
//   ACC[j] = sum_{k<j} sum_{o<4} z[k,j,o] * act_o(h_k @ W[k->j,o] + b[k->j,o])
// (op 4 is NullOp -> activation == 0 -> its GEMM is skipped).
// All edges leaving DAG level i are batched into one launch via blockIdx.z.
// A tiles are DMA'd global->LDS with GLOBAL_LOAD_ASYNC_TO_LDS_B128 (ASYNCcnt);
// B tiles (f32 weights, each read once from HBM) are converted to bf16 on the
// fly and stored transposed [n][k] with packed b32 DS stores.
// ---------------------------------------------------------------------------

#define N_CELLS 6
#define HIDDEN  1024
#define IN_DIM  376
#define IN_PAD  384
#define OUT_DIM 17
#define BATCH   2048

typedef __attribute__((ext_vector_type(16))) __bf16         v16bf;
typedef __attribute__((ext_vector_type(16))) unsigned short v16u;
typedef __attribute__((ext_vector_type(8)))  unsigned short v8u;
typedef __attribute__((ext_vector_type(8)))  float          v8f;

__device__ __forceinline__ unsigned short f2b(float f) {
    // round-to-nearest-even fp32 -> bf16
    unsigned int u = __builtin_bit_cast(unsigned int, f);
    unsigned int r = u + 0x7FFFu + ((u >> 16) & 1u);
    return (unsigned short)(r >> 16);
}

// -------------------- tiny kernels --------------------

__global__ void z_kernel(const float* __restrict__ la, const float* __restrict__ eps,
                         float* __restrict__ z) {
    int t = threadIdx.x;               // one row of 5 ops per thread; 30 rows
    if (t >= 30) return;
    float v[5]; float mx = -1e30f;
    #pragma unroll
    for (int o = 0; o < 5; ++o) { v[o] = la[t * 5 + o] + eps[t * 5 + o]; mx = fmaxf(mx, v[o]); }
    float s = 0.f;
    #pragma unroll
    for (int o = 0; o < 5; ++o) { v[o] = __expf(v[o] - mx); s += v[o]; }
    float inv = 1.f / s;
    #pragma unroll
    for (int o = 0; o < 5; ++o) z[t * 5 + o] = v[o] * inv;
}

__global__ void padx_kernel(const float* __restrict__ x, unsigned short* __restrict__ xp) {
    int idx = blockIdx.x * blockDim.x + threadIdx.x;
    if (idx >= BATCH * IN_PAD) return;
    int r = idx / IN_PAD, c = idx - r * IN_PAD;
    float v = (c < IN_DIM) ? x[r * IN_DIM + c] : 0.f;
    xp[idx] = f2b(v);
}

__global__ void cvt_kernel(const float* __restrict__ src, unsigned short* __restrict__ dst, int n) {
    int idx = blockIdx.x * blockDim.x + threadIdx.x;
    if (idx < n) dst[idx] = f2b(src[idx]);
}

__global__ void head_kernel(const float* __restrict__ acc, const float* __restrict__ Wout,
                            const float* __restrict__ bout, float* __restrict__ out) {
    int idx = blockIdx.x * blockDim.x + threadIdx.x;
    if (idx >= BATCH * OUT_DIM) return;
    int b = idx / OUT_DIM, j = idx - b * OUT_DIM;
    const float* row = acc + (size_t)b * HIDDEN;
    float s = 0.f;
    #pragma unroll 8
    for (int h = 0; h < HIDDEN; ++h) s += row[h] * Wout[h * OUT_DIM + j];
    out[idx] = tanhf(s + bout[j]);   // MAX_OUTPUT == 1.0
}

// -------------------- fused GEMM + bias + activation + z-mix --------------------
// One DAG level per launch; blockIdx.z enumerates target nodes j.
// A:    bf16 [BATCH x Kpad], zero-padded in K, row-major (shared by all j)
// B:    f32  [Ksrc x 1024] per (j, op); op stride Ksrc*1024, j stride jStrideB
// bias: f32  [n_ops x 1024] per j (j stride 5*1024)
// zrow: f32  [5] per j (j stride 5)
// Cacc: f32  [BATCH x 1024] per j; beta==0 -> overwrite, beta==1 -> accumulate

__global__ __launch_bounds__(256)
void gemm_mix_kernel(const unsigned short* __restrict__ A,
                     const float* __restrict__ B,
                     const float* __restrict__ bias,
                     const float* __restrict__ zrow,
                     float* __restrict__ Cacc,
                     int Kpad, int Ksrc, int beta, size_t jStrideB) {
    __shared__ __align__(16) unsigned short Alds[128][40];      // [m][k] bf16
    __shared__ __align__(16) unsigned short Blds[4][128][40];   // per-op [n][k] bf16 (transposed)

    const int tid   = threadIdx.x;
    const int wave  = tid >> 5;
    const int lane  = tid & 31;
    const int wm    = wave >> 2;       // 0..1 -> 64-row stripe
    const int wn    = wave & 3;        // 0..3 -> 32-col stripe
    const int lrow  = lane & 15;
    const int lhalf = lane >> 4;

    const int bz = blockIdx.z;
    B    += (size_t)bz * jStrideB;
    bias += (size_t)bz * 5 * HIDDEN;
    zrow += (size_t)bz * 5;
    Cacc += (size_t)bz * BATCH * HIDDEN;

    const int m0 = blockIdx.y * 128;
    const int n0 = blockIdx.x * 128;
    const size_t opStrideB = (size_t)Ksrc * HIDDEN;

    v8f c[4][4][2];                    // [op][mi][ni]
    #pragma unroll
    for (int o = 0; o < 4; ++o)
        #pragma unroll
        for (int mi = 0; mi < 4; ++mi)
            #pragma unroll
            for (int ni = 0; ni < 2; ++ni)
                c[o][mi][ni] = (v8f)0.f;

    // staging thread mappings (fixed across K iterations)
    const int ar   = tid >> 1;                 // A: row 0..127
    const int ac   = (tid & 1) * 16;           // A: 16-elem (32B) half of the 32-wide slice
    const int kl2  = tid >> 4;                 // B: k-pair 0..15 -> k = 2*kl2
    const int cseg = tid & 15;                 // B: 8 columns each
    const unsigned aLdsAddr = (unsigned)(size_t)&Alds[ar][ac];

    for (int kt = 0; kt < Kpad; kt += 32) {
        // ---- A tile (128x32 bf16): async DMA global -> LDS, overlaps B conversion
        {
            const unsigned short* ag = A + (size_t)(m0 + ar) * Kpad + kt + ac;
            asm volatile("global_load_async_to_lds_b128 %0, %1, off"
                         :: "v"(aLdsAddr), "v"(ag) : "memory");
            asm volatile("global_load_async_to_lds_b128 %0, %1, off offset:16"
                         :: "v"(aLdsAddr), "v"(ag) : "memory");
        }
        // ---- B tiles for 4 ops (32x128 f32 -> bf16, transposed, packed b32 stores)
        {
            const int k0 = kt + 2 * kl2;
            #pragma unroll
            for (int o = 0; o < 4; ++o) {
                const float* bp = B + (size_t)o * opStrideB + (size_t)k0 * HIDDEN + n0 + cseg * 8;
                float r0[8], r1[8];
                if (k0 < Ksrc) {
                    const float4* g = (const float4*)bp;
                    float4 f0 = g[0], f1 = g[1];
                    r0[0] = f0.x; r0[1] = f0.y; r0[2] = f0.z; r0[3] = f0.w;
                    r0[4] = f1.x; r0[5] = f1.y; r0[6] = f1.z; r0[7] = f1.w;
                } else {
                    #pragma unroll
                    for (int q = 0; q < 8; ++q) r0[q] = 0.f;
                }
                if (k0 + 1 < Ksrc) {
                    const float4* g = (const float4*)(bp + HIDDEN);
                    float4 f0 = g[0], f1 = g[1];
                    r1[0] = f0.x; r1[1] = f0.y; r1[2] = f0.z; r1[3] = f0.w;
                    r1[4] = f1.x; r1[5] = f1.y; r1[6] = f1.z; r1[7] = f1.w;
                } else {
                    #pragma unroll
                    for (int q = 0; q < 8; ++q) r1[q] = 0.f;
                }
                #pragma unroll
                for (int q = 0; q < 8; ++q) {
                    unsigned p = (unsigned)f2b(r0[q]) | ((unsigned)f2b(r1[q]) << 16);
                    *(unsigned*)&Blds[o][cseg * 8 + q][2 * kl2] = p;
                }
            }
        }
        asm volatile("s_wait_asynccnt 0x0" ::: "memory");
        __syncthreads();

        // ---- A fragments (ISA 16-bit A 16x32 layout): 4 per wave
        v16bf af[4];
        #pragma unroll
        for (int mi = 0; mi < 4; ++mi) {
            const unsigned short* ap = &Alds[wm * 64 + mi * 16 + lrow][lhalf * 8];
            v8u lo = *(const v8u*)ap;
            v8u hi = *(const v8u*)(ap + 16);
            v16u t;
            #pragma unroll
            for (int q = 0; q < 8; ++q) { t[q] = lo[q]; t[q + 8] = hi[q]; }
            af[mi] = __builtin_bit_cast(v16bf, t);
        }
        // ---- per op: B fragments (B 32x16 layout) + 8 WMMAs
        #pragma unroll
        for (int o = 0; o < 4; ++o) {
            v16bf bf[2];
            #pragma unroll
            for (int ni = 0; ni < 2; ++ni) {
                const unsigned short* bp = &Blds[o][wn * 32 + ni * 16 + lrow][lhalf * 8];
                v8u lo = *(const v8u*)bp;
                v8u hi = *(const v8u*)(bp + 16);
                v16u t;
                #pragma unroll
                for (int q = 0; q < 8; ++q) { t[q] = lo[q]; t[q + 8] = hi[q]; }
                bf[ni] = __builtin_bit_cast(v16bf, t);
            }
            #pragma unroll
            for (int mi = 0; mi < 4; ++mi)
                #pragma unroll
                for (int ni = 0; ni < 2; ++ni)
                    c[o][mi][ni] = __builtin_amdgcn_wmma_f32_16x16x32_bf16(
                        false, af[mi], false, bf[ni], (short)0, c[o][mi][ni], false, false);
        }
        __syncthreads();
    }

    // ---- bias + activation + z-weighted mix over ops
    v8f mix[4][2];
    #pragma unroll
    for (int mi = 0; mi < 4; ++mi)
        #pragma unroll
        for (int ni = 0; ni < 2; ++ni)
            mix[mi][ni] = (v8f)0.f;

    #pragma unroll
    for (int o = 0; o < 4; ++o) {
        const float zo = zrow[o];
        #pragma unroll
        for (int ni = 0; ni < 2; ++ni) {
            const int n = n0 + wn * 32 + ni * 16 + lrow;
            const float bv = bias[o * HIDDEN + n];
            #pragma unroll
            for (int mi = 0; mi < 4; ++mi) {
                #pragma unroll
                for (int q = 0; q < 8; ++q) {
                    float pre = c[o][mi][ni][q] + bv;
                    float a;
                    if      (o == 0) a = tanhf(pre);
                    else if (o == 1) a = fmaxf(pre, 0.f);
                    else if (o == 2) a = (pre > 0.f) ? pre : (__expf(pre) - 1.f);
                    else             a = (pre > 0.f) ? pre : 0.01f * pre;
                    mix[mi][ni][q] += zo * a;
                }
            }
        }
    }

    // ---- write node accumulator (each element owned by exactly one lane)
    if (beta) {
        #pragma unroll
        for (int mi = 0; mi < 4; ++mi)
            #pragma unroll
            for (int ni = 0; ni < 2; ++ni) {
                const int n = n0 + wn * 32 + ni * 16 + lrow;
                #pragma unroll
                for (int q = 0; q < 8; ++q) {
                    const int m = m0 + wm * 64 + mi * 16 + q + 8 * lhalf;
                    const size_t idx = (size_t)m * HIDDEN + n;
                    Cacc[idx] += mix[mi][ni][q];
                }
            }
    } else {
        #pragma unroll
        for (int mi = 0; mi < 4; ++mi)
            #pragma unroll
            for (int ni = 0; ni < 2; ++ni) {
                const int n = n0 + wn * 32 + ni * 16 + lrow;
                #pragma unroll
                for (int q = 0; q < 8; ++q) {
                    const int m = m0 + wm * 64 + mi * 16 + q + 8 * lhalf;
                    const size_t idx = (size_t)m * HIDDEN + n;
                    Cacc[idx] = mix[mi][ni][q];
                }
            }
    }
}

// -------------------- launcher --------------------

extern "C" void kernel_launch(void* const* d_in, const int* in_sizes, int n_in,
                              void* d_out, int out_size, void* d_ws, size_t ws_size,
                              hipStream_t stream) {
    const float* x    = (const float*)d_in[0];
    const float* W0   = (const float*)d_in[1];   // [6,5,376,1024]
    const float* b0   = (const float*)d_in[2];   // [6,5,1024]
    const float* Wh   = (const float*)d_in[3];   // [4,6,5,1024,1024]
    const float* bh   = (const float*)d_in[4];   // [4,6,5,1024]
    const float* Wout = (const float*)d_in[5];
    const float* bout = (const float*)d_in[6];
    const float* la   = (const float*)d_in[7];   // [5,6,5]
    const float* eps  = (const float*)d_in[8];

    char* ws = (char*)d_ws;
    float*          z   = (float*)ws;                                          // 150 f32
    unsigned short* xp  = (unsigned short*)(ws + 1024);                        // 2048*384 bf16
    unsigned short* hb  = (unsigned short*)(ws + 1024 + (size_t)BATCH * IN_PAD * 2);
    float*          acc = (float*)(ws + 1024 + (size_t)BATCH * IN_PAD * 2
                                          + (size_t)BATCH * HIDDEN * 2);       // 5 x [2048x1024]
    const size_t MAT = (size_t)BATCH * HIDDEN;

    z_kernel<<<1, 32, 0, stream>>>(la, eps, z);
    padx_kernel<<<(BATCH * IN_PAD + 255) / 256, 256, 0, stream>>>(x, xp);

    dim3 blk(256);

    // level 0: cell 0 -> nodes j = 1..5 (batched over blockIdx.z)
    {
        dim3 grid(HIDDEN / 128, BATCH / 128, N_CELLS - 1);
        gemm_mix_kernel<<<grid, blk, 0, stream>>>(
            xp,
            W0 + (size_t)1 * 5 * IN_DIM * HIDDEN,           // j = 1 base
            b0 + (size_t)1 * 5 * HIDDEN,
            z  + (size_t)(0 * N_CELLS + 1) * 5,
            acc,                                            // ACC[1] base
            IN_PAD, IN_DIM, /*beta=*/0,
            (size_t)5 * IN_DIM * HIDDEN);
    }

    // levels i = 1..4: node i -> nodes j = i+1..5 (batched over blockIdx.z)
    for (int i = 1; i < N_CELLS - 1; ++i) {
        cvt_kernel<<<((int)MAT + 255) / 256, 256, 0, stream>>>(
            acc + (size_t)(i - 1) * MAT, hb, (int)MAT);
        dim3 grid(HIDDEN / 128, BATCH / 128, N_CELLS - 1 - i);
        gemm_mix_kernel<<<grid, blk, 0, stream>>>(
            hb,
            Wh + ((size_t)(i - 1) * N_CELLS + (i + 1)) * 5 * HIDDEN * HIDDEN,  // j = i+1 base
            bh + ((size_t)(i - 1) * N_CELLS + (i + 1)) * 5 * HIDDEN,
            z  + (size_t)(i * N_CELLS + (i + 1)) * 5,
            acc + (size_t)i * MAT,                                             // ACC[i+1] base
            HIDDEN, HIDDEN, /*beta=*/1,
            (size_t)5 * HIDDEN * HIDDEN);
    }

    head_kernel<<<(BATCH * OUT_DIM + 255) / 256, 256, 0, stream>>>(
        acc + 4 * MAT, Wout, bout, (float*)d_out);
}
